// StressNNEval_21921513079376
// MI455X (gfx1250) — matrix operand order repaired and checked by reading, not compiled, not run on hardware
//
#include <hip/hip_runtime.h>
#include <hip/hip_bf16.h>

typedef _Float16 v16h __attribute__((ext_vector_type(16)));
typedef _Float16 h8   __attribute__((ext_vector_type(8)));
typedef float    v8f  __attribute__((ext_vector_type(8)));

#define ACT_LD 264              // halves per activation row (padded vs 256: bank-conflict avoidance)
#define ROWS   256              // rows per block (2 M-tiles per wave, 8 waves)
#define WBUF_BYTES 131072       // two 64KB half-buffers (pipelined weight staging)
#define ACT_BYTES  (ROWS * ACT_LD * 2)     // 135168
#define Y5_BYTES   (ROWS * 16 * 4)         // 16384
#define RSM_BYTES  (ROWS * 9 * 4)          // 9216
#define SMEM_BYTES (WBUF_BYTES + ACT_BYTES + Y5_BYTES + RSM_BYTES)   // 291840 < 320K

// workspace offsets (bytes)
#define WS_B1P  0
#define WS_W1S  1024
#define WS_W2S  17408
#define WS_W3S  148480
#define WS_W4S  279552
#define WS_W5S  410624

// ---------------------------------------------------------------------------
// Setup kernel 1: fold constant latent (emb[traj] @ W1[25:153,:]) into bias1.
// ---------------------------------------------------------------------------
__global__ void b1p_kernel(const float* __restrict__ W1, const float* __restrict__ b1,
                           const float* __restrict__ emb, const int* __restrict__ traj,
                           float* __restrict__ b1p) {
    int n = threadIdx.x;            // 256 threads
    int t = traj[0];
    float s = b1[n];
    for (int k = 0; k < 128; ++k)
        s += emb[t * 128 + k] * W1[(25 + k) * 256 + n];
    b1p[n] = s;
}

// ---------------------------------------------------------------------------
// Setup kernel 2: swizzle f32 weights [K x N] into per-lane WMMA B-fragment
// f16 layout, *nt-major* fragment order (frag = nt*nKt + kt) so each
// half-layer (nt 0..7 / 8..15) is one contiguous 64KB slice for pipelining.
// Lane l holds 16 contiguous halves: element e -> K = kt*32 + (l>>4)*8 + (e&7) + (e>>3)*16,
// N = nt*16 + (l&15).
// ---------------------------------------------------------------------------
__global__ void swizzle_kernel(const float* __restrict__ W, _Float16* __restrict__ dst,
                               int Kreal, int Nreal, int ldn, int nKt, int nNt) {
    int idx = blockIdx.x * 256 + threadIdx.x;
    int total = nKt * nNt * 512;
    if (idx >= total) return;
    int e    = idx & 15;
    int lane = (idx >> 4) & 31;
    int frag = idx >> 9;
    int kt = frag % nKt;            // nt-major
    int nt = frag / nKt;
    int k = kt * 32 + ((lane >> 4) * 8) + (e & 7) + ((e >> 3) * 16);
    int n = nt * 16 + (lane & 15);
    float v = (k < Kreal && n < Nreal) ? W[k * ldn + n] : 0.0f;
    dst[idx] = (_Float16)v;
}

// ---------------------------------------------------------------------------
// Fragment loaders
// ---------------------------------------------------------------------------
// A fragment (16x32 f16): lane l = row (l&15); K runs { kb..kb+7, kb+16..kb+23 }
// with kb = (l>>4)*8, per ISA 7.12.2 "16-bit A-Matrix 16x32".
__device__ __forceinline__ v16h load_a_frag(const _Float16* rowp, int kbase) {
    h8 lo = *(const h8*)(rowp + kbase);
    h8 hi = *(const h8*)(rowp + kbase + 16);
    return __builtin_shufflevector(lo, hi, 0, 1, 2, 3, 4, 5, 6, 7,
                                   8, 9, 10, 11, 12, 13, 14, 15);
}

// B fragment: pre-swizzled, contiguous 32B per lane.
__device__ __forceinline__ v16h load_b_frag(const _Float16* wb, int frag, int lane) {
    const h8* p = (const h8*)(wb + (size_t)frag * 512 + lane * 16);
    h8 lo = p[0];
    h8 hi = p[1];
    return __builtin_shufflevector(lo, hi, 0, 1, 2, 3, 4, 5, 6, 7,
                                   8, 9, 10, 11, 12, 13, 14, 15);
}

// Branchless exact-GELU: erf via Abramowitz-Stegun 7.1.26 (|err|<=1.5e-7,
// << f16 activation quantization), using hardware v_rcp_f32 / v_exp_f32.
__device__ __forceinline__ float gelu_fast(float x) {
    float z    = 0.70710678118654752f * x;
    float az   = fabsf(z);
    float t    = __builtin_amdgcn_rcpf(fmaf(0.3275911f, az, 1.0f));
    float poly = fmaf(fmaf(fmaf(fmaf(1.061405429f, t, -1.453152027f),
                                t, 1.421413741f),
                           t, -0.284496736f),
                      t, 0.254829592f) * t;
    float e    = __builtin_amdgcn_exp2f(-az * az * 1.4426950408889634f);
    float erfv = fmaf(-poly, e, 1.0f);           // erf(|z|)
    return 0.5f * x * (1.0f + copysignf(erfv, x));
}

// ---------------------------------------------------------------------------
// Fused kernel: polar/SVD features -> 5-layer MLP (WMMA f16) -> stress assembly
// 256 threads = 8 waves; 256 rows per block (2 x 16-row M-tiles per wave).
// Weight staging is software-pipelined with compute at half-layer granularity.
// ---------------------------------------------------------------------------
__global__ __launch_bounds__(256, 1)
void stress_mlp_kernel(const float* __restrict__ Fm, const float* __restrict__ Cm,
                       const float* __restrict__ b2, const float* __restrict__ b3,
                       const float* __restrict__ b4, const float* __restrict__ b5,
                       const float* __restrict__ b1p,
                       const _Float16* __restrict__ W1s, const _Float16* __restrict__ W2s,
                       const _Float16* __restrict__ W3s, const _Float16* __restrict__ W4s,
                       const _Float16* __restrict__ W5s,
                       float* __restrict__ out) {
    extern __shared__ char smem[];
    _Float16* wbuf = (_Float16*)smem;                                // 2 x 64KB halves
    _Float16* act  = (_Float16*)(smem + WBUF_BYTES);                 // activations (f16)
    float*    y5   = (float*)(smem + WBUF_BYTES + ACT_BYTES);        // layer-5 staging
    float*    rsm  = (float*)(smem + WBUF_BYTES + ACT_BYTES + Y5_BYTES); // rotation R

    uint4* wbufA = (uint4*)wbuf;            // bytes [0, 64K)
    uint4* wbufB = wbufA + 4096;            // bytes [64K, 128K)

    const int tid  = threadIdx.x;
    const int lane = tid & 31;
    const int wv   = tid >> 5;
    const int base = blockIdx.x * ROWS;

    // ---- stage 0: W1 -> start of halfB; preprocess one element/thread -----
    {
        const uint4* s = (const uint4*)W1s;
        for (int i = tid; i < 1024; i += 256) wbufB[i] = s[i];
    }

    {
        int gid = base + tid;
        const float* fp = Fm + gid * 9;
        const float* cp = Cm + gid * 9;
        float f00 = fp[0], f01 = fp[1], f02 = fp[2];
        float f10 = fp[3], f11 = fp[4], f12 = fp[5];
        float f20 = fp[6], f21 = fp[7], f22 = fp[8];

        // A = F^T F (symmetric)
        float a00 = f00 * f00 + f10 * f10 + f20 * f20;
        float a01 = f00 * f01 + f10 * f11 + f20 * f21;
        float a02 = f00 * f02 + f10 * f12 + f20 * f22;
        float a11 = f01 * f01 + f11 * f11 + f21 * f21;
        float a12 = f01 * f02 + f11 * f12 + f21 * f22;
        float a22 = f02 * f02 + f12 * f12 + f22 * f22;
        float det = f00 * (f11 * f22 - f12 * f21)
                  - f01 * (f10 * f22 - f12 * f20)
                  + f02 * (f10 * f21 - f11 * f20);

        // closed-form eigenvalues of A (descending) -> singular values of F
        float q  = (a00 + a11 + a22) * (1.0f / 3.0f);
        float p1 = a01 * a01 + a02 * a02 + a12 * a12;
        float d0 = a00 - q, d1 = a11 - q, d2 = a22 - q;
        float p2 = d0 * d0 + d1 * d1 + d2 * d2 + 2.0f * p1;
        float pp = sqrtf(fmaxf(p2 * (1.0f / 6.0f), 1e-30f));
        float ip = 1.0f / pp;
        float c00 = d0 * ip, c01 = a01 * ip, c02 = a02 * ip;
        float c11 = d1 * ip, c12 = a12 * ip, c22 = d2 * ip;
        float detB = c00 * (c11 * c22 - c12 * c12)
                   - c01 * (c01 * c22 - c12 * c02)
                   + c02 * (c01 * c12 - c11 * c02);
        float rc  = fminf(fmaxf(detB * 0.5f, -1.0f), 1.0f);
        float phi = acosf(rc) * (1.0f / 3.0f);
        float e1 = q + 2.0f * pp * cosf(phi);
        float e3 = q + 2.0f * pp * cosf(phi + 2.09439510239319549f);
        float e2 = 3.0f * q - e1 - e3;
        float sg0 = sqrtf(fmaxf(e1, 0.0f));
        float sg1 = sqrtf(fmaxf(e2, 0.0f));
        float sg2 = sqrtf(fmaxf(e3, 0.0f));

        // polar rotation via Newton: X <- 0.5*(X + X^{-T}); X^{-T} = cof(X)/det(X)
        float x00 = f00, x01 = f01, x02 = f02;
        float x10 = f10, x11 = f11, x12 = f12;
        float x20 = f20, x21 = f21, x22 = f22;
#pragma unroll
        for (int it = 0; it < 6; ++it) {
            float k00 =  x11 * x22 - x12 * x21;
            float k01 = -(x10 * x22 - x12 * x20);
            float k02 =  x10 * x21 - x11 * x20;
            float k10 = -(x01 * x22 - x02 * x21);
            float k11 =  x00 * x22 - x02 * x20;
            float k12 = -(x00 * x21 - x01 * x20);
            float k20 =  x01 * x12 - x02 * x11;
            float k21 = -(x00 * x12 - x02 * x10);
            float k22 =  x00 * x11 - x01 * x10;
            float dx  = x00 * k00 + x01 * k01 + x02 * k02;
            float hf  = 0.5f / dx;
            x00 = 0.5f * x00 + k00 * hf;  x01 = 0.5f * x01 + k01 * hf;  x02 = 0.5f * x02 + k02 * hf;
            x10 = 0.5f * x10 + k10 * hf;  x11 = 0.5f * x11 + k11 * hf;  x12 = 0.5f * x12 + k12 * hf;
            x20 = 0.5f * x20 + k20 * hf;  x21 = 0.5f * x21 + k21 * hf;  x22 = 0.5f * x22 + k22 * hf;
        }
        float* rdst = rsm + tid * 9;
        rdst[0] = x00; rdst[1] = x01; rdst[2] = x02;
        rdst[3] = x10; rdst[4] = x11; rdst[5] = x12;
        rdst[6] = x20; rdst[7] = x21; rdst[8] = x22;

        // feature row (25 valid halves, zero-padded to 32)
        float f00c = fmaxf(f00, 1e-6f);
        _Float16* arow = act + tid * ACT_LD;
        arow[0]  = (_Float16)(sg0 - 1.0f);
        arow[1]  = (_Float16)(sg1 - 1.0f);
        arow[2]  = (_Float16)(sg2 - 1.0f);
        arow[3]  = (_Float16)(a00 - 1.0f);
        arow[4]  = (_Float16)a01;
        arow[5]  = (_Float16)a02;
        arow[6]  = (_Float16)a01;
        arow[7]  = (_Float16)(a11 - 1.0f);
        arow[8]  = (_Float16)a12;
        arow[9]  = (_Float16)a02;
        arow[10] = (_Float16)a12;
        arow[11] = (_Float16)(a22 - 1.0f);
        arow[12] = (_Float16)(det - 1.0f);
        arow[13] = (_Float16)(logf(det) - 1.0f);
        arow[14] = (_Float16)(f00c - 1.0f);
        arow[15] = (_Float16)(logf(f00c) - 1.0f);
#pragma unroll
        for (int i = 0; i < 9; ++i) arow[16 + i] = (_Float16)cp[i];
#pragma unroll
        for (int i = 25; i < 32; ++i) arow[i] = (_Float16)0.0f;
    }
    __syncthreads();

    // per-wave geometry: wave owns rows [wv*32, wv*32+32) = two 16-row M-tiles
    const _Float16* rowA0 = act + (wv * 32 + (lane & 15)) * ACT_LD;
    const _Float16* rowA1 = rowA0 + 16 * ACT_LD;
    const int khA = (lane >> 4) * 8;    // A-fragment K sub-base per lane group
    const int nco = lane & 15;          // D: output column within N-tile
    const int r0  = wv * 32 + (lane >> 4) * 8;   // D: output row base, tile 0

    // ---- layer 1: K=32, N=256, reads W1 region (inside halfB, offset 32768h);
    //      concurrently streams W2-half0 (4096 uint4) into halfA --------------
    {
        const _Float16* w1b = wbuf + 32768;     // byte offset 65536
        const uint4* csrc = (const uint4*)W2s;
        v16h a0 = load_a_frag(rowA0, khA);
        v16h a1 = load_a_frag(rowA1, khA);
        for (int nt = 0; nt < 16; ++nt) {
            int ci = nt * 256 + tid;            // 1 uint4 per thread per iter
            uint4 t0 = csrc[ci];
            v16h bv = load_b_frag(w1b, nt, lane);
            float bb = b1p[nt * 16 + nco];
            v8f acc0 = {}, acc1 = {};
            acc0 = __builtin_amdgcn_wmma_f32_16x16x32_f16(false, a0, false, bv,
                                                          (short)0, acc0, false, false);
            acc1 = __builtin_amdgcn_wmma_f32_16x16x32_f16(false, a1, false, bv,
                                                          (short)0, acc1, false, false);
            wbufA[ci] = t0;
#pragma unroll
            for (int v = 0; v < 8; ++v) {
                act[(r0 + v) * ACT_LD + nt * 16 + nco]      = (_Float16)gelu_fast(acc0[v] + bb);
                act[(r0 + 16 + v) * ACT_LD + nt * 16 + nco] = (_Float16)gelu_fast(acc1[v] + bb);
            }
        }
    }
    __syncthreads();

    // ---- pipelined dense 256x256 layer with GELU ----------------------------
    // nt 0..7  : compute from halfA, stream cp1 (this layer's half1) -> halfB
    // nt 8..15 : compute from halfB, stream cp2 (next layer's half0) -> halfA
    auto dense_layer = [&](const float* bias,
                           const uint4* cp1_src, const uint4* cp2_src, int cp2_n) {
        v16h a0[8], a1[8];
#pragma unroll
        for (int kt = 0; kt < 8; ++kt) {
            a0[kt] = load_a_frag(rowA0, kt * 32 + khA);
            a1[kt] = load_a_frag(rowA1, kt * 32 + khA);
        }
        for (int nt = 0; nt < 16; ++nt) {
            if (nt == 8) __syncthreads();       // half1 staged; halfA now free
            const uint4* csrc;
            uint4* cdst;
            int cn, ci;
            if (nt < 8) { csrc = cp1_src; cdst = wbufB; cn = 4096; ci = nt; }
            else        { csrc = cp2_src; cdst = wbufA; cn = cp2_n; ci = nt - 8; }
            int i0 = ci * 512 + tid;            // 2 uint4 per thread per iter
            bool d0c = i0 < cn, d1c = (i0 + 256) < cn;   // wave-uniform
            uint4 t0, t1;
            if (d0c) t0 = csrc[i0];
            if (d1c) t1 = csrc[i0 + 256];

            float bb = bias[nt * 16 + nco];
            v8f acc0 = {}, acc1 = {};
#pragma unroll
            for (int kt = 0; kt < 8; ++kt) {
                v16h bv = load_b_frag(wbuf, nt * 8 + kt, lane);
                acc0 = __builtin_amdgcn_wmma_f32_16x16x32_f16(false, a0[kt], false, bv,
                                                              (short)0, acc0, false, false);
                acc1 = __builtin_amdgcn_wmma_f32_16x16x32_f16(false, a1[kt], false, bv,
                                                              (short)0, acc1, false, false);
            }
            if (d0c) cdst[i0] = t0;
            if (d1c) cdst[i0 + 256] = t1;
#pragma unroll
            for (int v = 0; v < 8; ++v) {
                act[(r0 + v) * ACT_LD + nt * 16 + nco]      = (_Float16)gelu_fast(acc0[v] + bb);
                act[(r0 + 16 + v) * ACT_LD + nt * 16 + nco] = (_Float16)gelu_fast(acc1[v] + bb);
            }
        }
        __syncthreads();
    };

    dense_layer(b2, (const uint4*)W2s + 4096, (const uint4*)W3s, 4096);
    dense_layer(b3, (const uint4*)W3s + 4096, (const uint4*)W4s, 4096);
    dense_layer(b4, (const uint4*)W4s + 4096, (const uint4*)W5s, 512);

    // ---- layer 5: K=256 (8 ktiles in halfA), N=16 (9 valid), no GELU -------
    {
        v16h a0[8], a1[8];
#pragma unroll
        for (int kt = 0; kt < 8; ++kt) {
            a0[kt] = load_a_frag(rowA0, kt * 32 + khA);
            a1[kt] = load_a_frag(rowA1, kt * 32 + khA);
        }
        v8f acc0 = {}, acc1 = {};
#pragma unroll
        for (int kt = 0; kt < 8; ++kt) {
            v16h bv = load_b_frag(wbuf, kt, lane);
            acc0 = __builtin_amdgcn_wmma_f32_16x16x32_f16(false, a0[kt], false, bv,
                                                          (short)0, acc0, false, false);
            acc1 = __builtin_amdgcn_wmma_f32_16x16x32_f16(false, a1[kt], false, bv,
                                                          (short)0, acc1, false, false);
        }
        if (nco < 9) {
            float bb = b5[nco];
#pragma unroll
            for (int v = 0; v < 8; ++v) {
                y5[(r0 + v) * 16 + nco]      = acc0[v] + bb;
                y5[(r0 + 16 + v) * 16 + nco] = acc1[v] + bb;
            }
        }
    }
    __syncthreads();

    // ---- final: symmetrize, P = R*S, cauchy = P * F^T ----------------------
    {
        int gid = base + tid;
        const float* fp = Fm + gid * 9;
        const float* rr = rsm + tid * 9;
        const float* yy = y5 + tid * 16;
        float x00 = yy[0], x01 = yy[1], x02 = yy[2];
        float x10 = yy[3], x11 = yy[4], x12 = yy[5];
        float x20 = yy[6], x21 = yy[7], x22 = yy[8];
        float s00 = x00, s11 = x11, s22 = x22;
        float s01 = 0.5f * (x01 + x10);
        float s02 = 0.5f * (x02 + x20);
        float s12 = 0.5f * (x12 + x21);
        float r00 = rr[0], r01 = rr[1], r02 = rr[2];
        float r10 = rr[3], r11 = rr[4], r12 = rr[5];
        float r20 = rr[6], r21 = rr[7], r22 = rr[8];
        float p00 = r00 * s00 + r01 * s01 + r02 * s02;
        float p01 = r00 * s01 + r01 * s11 + r02 * s12;
        float p02 = r00 * s02 + r01 * s12 + r02 * s22;
        float p10 = r10 * s00 + r11 * s01 + r12 * s02;
        float p11 = r10 * s01 + r11 * s11 + r12 * s12;
        float p12 = r10 * s02 + r11 * s12 + r12 * s22;
        float p20 = r20 * s00 + r21 * s01 + r22 * s02;
        float p21 = r20 * s01 + r21 * s11 + r22 * s12;
        float p22 = r20 * s02 + r21 * s12 + r22 * s22;
        float f00 = fp[0], f01 = fp[1], f02 = fp[2];
        float f10 = fp[3], f11 = fp[4], f12 = fp[5];
        float f20 = fp[6], f21 = fp[7], f22 = fp[8];
        float* op = out + gid * 9;
        op[0] = p00 * f00 + p01 * f01 + p02 * f02;
        op[1] = p00 * f10 + p01 * f11 + p02 * f12;
        op[2] = p00 * f20 + p01 * f21 + p02 * f22;
        op[3] = p10 * f00 + p11 * f01 + p12 * f02;
        op[4] = p10 * f10 + p11 * f11 + p12 * f12;
        op[5] = p10 * f20 + p11 * f21 + p12 * f22;
        op[6] = p20 * f00 + p21 * f01 + p22 * f02;
        op[7] = p20 * f10 + p21 * f11 + p22 * f12;
        op[8] = p20 * f20 + p21 * f21 + p22 * f22;
    }
}

// ---------------------------------------------------------------------------
extern "C" void kernel_launch(void* const* d_in, const int* in_sizes, int n_in,
                              void* d_out, int out_size, void* d_ws, size_t ws_size,
                              hipStream_t stream) {
    const float* F    = (const float*)d_in[0];
    const float* C    = (const float*)d_in[1];
    const float* emb  = (const float*)d_in[2];
    const int*   traj = (const int*)d_in[3];
    const float* W1 = (const float*)d_in[4];
    const float* b1 = (const float*)d_in[5];
    const float* W2 = (const float*)d_in[6];
    const float* b2 = (const float*)d_in[7];
    const float* W3 = (const float*)d_in[8];
    const float* b3 = (const float*)d_in[9];
    const float* W4 = (const float*)d_in[10];
    const float* b4 = (const float*)d_in[11];
    const float* W5 = (const float*)d_in[12];
    const float* b5 = (const float*)d_in[13];
    float* out = (float*)d_out;

    const int B = in_sizes[0] / 9;

    char* ws = (char*)d_ws;
    float*     b1p = (float*)(ws + WS_B1P);
    _Float16*  W1s = (_Float16*)(ws + WS_W1S);
    _Float16*  W2s = (_Float16*)(ws + WS_W2S);
    _Float16*  W3s = (_Float16*)(ws + WS_W3S);
    _Float16*  W4s = (_Float16*)(ws + WS_W4S);
    _Float16*  W5s = (_Float16*)(ws + WS_W5S);

    // setup: fold latent into bias1, pre-swizzle weights to f16 WMMA fragments
    b1p_kernel<<<1, 256, 0, stream>>>(W1, b1, emb, traj, b1p);
    swizzle_kernel<<<(1 * 16 * 512 + 255) / 256, 256, 0, stream>>>(W1, W1s, 25, 256, 256, 1, 16);
    swizzle_kernel<<<(8 * 16 * 512 + 255) / 256, 256, 0, stream>>>(W2, W2s, 256, 256, 256, 8, 16);
    swizzle_kernel<<<(8 * 16 * 512 + 255) / 256, 256, 0, stream>>>(W3, W3s, 256, 256, 256, 8, 16);
    swizzle_kernel<<<(8 * 16 * 512 + 255) / 256, 256, 0, stream>>>(W4, W4s, 256, 256, 256, 8, 16);
    swizzle_kernel<<<(8 * 1 * 512 + 255) / 256, 256, 0, stream>>>(W5, W5s, 256, 9, 9, 8, 1);

    (void)hipFuncSetAttribute((const void*)stress_mlp_kernel,
                              hipFuncAttributeMaxDynamicSharedMemorySize, SMEM_BYTES);
    stress_mlp_kernel<<<B / ROWS, 256, SMEM_BYTES, stream>>>(
        F, C, b2, b3, b4, b5, b1p, W1s, W2s, W3s, W4s, W5s, out);
}
